// GNNStack_26379689132032
// MI455X (gfx1250) — matrix operand order repaired and checked by reading, not compiled, not run on hardware
//
#include <hip/hip_runtime.h>

#define N_NODES 50000
#define N_EDGES 800000
#define D_HID 256
#define D_OUTC 128

typedef __attribute__((ext_vector_type(16))) __bf16 v16bf;
typedef __attribute__((ext_vector_type(8)))  float  v8f;

// ======================= CSR build =======================
__global__ void zero_i32_k(int* __restrict__ p, int n) {
  int i = blockIdx.x * 256 + threadIdx.x;
  if (i < n) p[i] = 0;
}

__global__ void hist_k(const long long* __restrict__ ei, int* __restrict__ cnt) {
  int e = blockIdx.x * 256 + threadIdx.x;
  if (e < N_EDGES) {
    int d = (int)ei[N_EDGES + e];
    atomicAdd(&cnt[d], 1);
  }
}

__global__ void scan_blocks_k(const int* __restrict__ cnt, int* __restrict__ excl,
                              int* __restrict__ bsum, int n) {
  __shared__ int sh[256];
  int tid = threadIdx.x;
  int i = blockIdx.x * 256 + tid;
  int v = (i < n) ? cnt[i] : 0;
  sh[tid] = v;
  __syncthreads();
  #pragma unroll
  for (int off = 1; off < 256; off <<= 1) {
    int t = (tid >= off) ? sh[tid - off] : 0;
    __syncthreads();
    sh[tid] += t;
    __syncthreads();
  }
  if (i < n) excl[i] = sh[tid] - v;
  if (tid == 255) bsum[blockIdx.x] = sh[255];
}

__global__ void scan_small_k(int* __restrict__ data, int n) {
  __shared__ int sh[256];
  int tid = threadIdx.x;
  int v = (tid < n) ? data[tid] : 0;
  sh[tid] = v;
  __syncthreads();
  #pragma unroll
  for (int off = 1; off < 256; off <<= 1) {
    int t = (tid >= off) ? sh[tid - off] : 0;
    __syncthreads();
    sh[tid] += t;
    __syncthreads();
  }
  if (tid < n) data[tid] = sh[tid] - v;
}

__global__ void add_off_k(const int* __restrict__ excl, const int* __restrict__ bsum,
                          int* __restrict__ row_ptr, int* __restrict__ cursor, int n) {
  int i = blockIdx.x * 256 + threadIdx.x;
  if (i < n) {
    int v = excl[i] + bsum[blockIdx.x];
    row_ptr[i] = v;
    cursor[i]  = v;
  }
}

__global__ void fill_k(const long long* __restrict__ ei, int* __restrict__ cursor,
                       int* __restrict__ col) {
  int e = blockIdx.x * 256 + threadIdx.x;
  if (e < N_EDGES) {
    int s = (int)ei[e];
    int d = (int)ei[N_EDGES + e];
    int slot = atomicAdd(&cursor[d], 1);
    col[slot] = s;
  }
}

// ======================= mean aggregation (CSR gather) =======================
// one wave per node; lane owns 8 contiguous features -> full 1KB row coalesced
__global__ void agg_mean_k(const float* __restrict__ x, const int* __restrict__ row_ptr,
                           const int* __restrict__ cnt, const int* __restrict__ col,
                           float* __restrict__ agg) {
  int node = blockIdx.x * 8 + (threadIdx.x >> 5);
  if (node >= N_NODES) return;
  int lane  = threadIdx.x & 31;
  int start = row_ptr[node];
  int deg   = cnt[node];
  float a0=0,a1=0,a2=0,a3=0,a4=0,a5=0,a6=0,a7=0;
  for (int i = 0; i < deg; ++i) {
    int s = col[start + i];
    const float4* p = (const float4*)(x + (size_t)s * D_HID + lane * 8);
    float4 u = p[0], v = p[1];
    a0 += u.x; a1 += u.y; a2 += u.z; a3 += u.w;
    a4 += v.x; a5 += v.y; a6 += v.z; a7 += v.w;
  }
  float inv = 1.0f / fmaxf((float)deg, 1.0f);
  float4* q = (float4*)(agg + (size_t)node * D_HID + lane * 8);
  q[0] = make_float4(a0*inv, a1*inv, a2*inv, a3*inv);
  q[1] = make_float4(a4*inv, a5*inv, a6*inv, a7*inv);
}

// ======================= weight packing (fp32 -> bf16 hi/lo in WMMA B-layout) ==
// per (k-tile, n-tile, lane): 16 hi bf16 then 16 lo bf16, contiguous per lane.
// B 32x16 layout: lanes 0-15 -> K={0..7,16..23}, lanes 16-31 -> K={8..15,24..31}.
__global__ void pack_w_k(const float* __restrict__ W, __bf16* __restrict__ out,
                         int K, int N) {
  int ntn = N >> 4;
  int total = (K >> 5) * ntn * 32;
  int idx = blockIdx.x * 256 + threadIdx.x;
  if (idx >= total) return;
  int lane = idx & 31;
  int tile = idx >> 5;
  int nt = tile % ntn;
  int kt = tile / ntn;
  int n  = (nt << 4) + (lane & 15);
  int kg = (lane < 16) ? 0 : 8;
  __bf16* dst = out + (size_t)idx * 32;
  #pragma unroll
  for (int e = 0; e < 16; ++e) {
    float v = W[(size_t)((kt << 5) + kg + (e < 8 ? e : e + 8)) * N + n];
    __bf16 h = (__bf16)v;
    dst[e]      = h;
    dst[16 + e] = (__bf16)(v - (float)h);
  }
}

// ======================= WMMA GEMM helpers =======================
// A 16x32 bf16 layout: lanes 0-15: K = {0..7, 16..23}; lanes 16-31: K = {8..15, 24..31}
__device__ __forceinline__ void load_a16(const float* __restrict__ arow, int koff,
                                         v16bf& ah, v16bf& al) {
  const float4* p0 = (const float4*)(arow + koff);
  const float4* p1 = (const float4*)(arow + koff + 16);
  float4 q0 = p0[0], q1 = p0[1], q2 = p1[0], q3 = p1[1];
  float f[16] = {q0.x,q0.y,q0.z,q0.w, q1.x,q1.y,q1.z,q1.w,
                 q2.x,q2.y,q2.z,q2.w, q3.x,q3.y,q3.z,q3.w};
  #pragma unroll
  for (int e = 0; e < 16; ++e) {
    __bf16 h = (__bf16)f[e];
    ah[e] = h;
    al[e] = (__bf16)(f[e] - (float)h);
  }
}

// error-compensated bf16 product: ah*bh + ah*bl + al*bh
__device__ __forceinline__ void wmma3(v8f& c, const v16bf& ah, const v16bf& al,
                                      const __bf16* __restrict__ bp) {
  v16bf bh = *(const v16bf*)bp;
  v16bf bl = *(const v16bf*)(bp + 16);
  c = __builtin_amdgcn_wmma_f32_16x16x32_bf16(false, ah, false, bh, (short)0, c, false, false);
  c = __builtin_amdgcn_wmma_f32_16x16x32_bf16(false, ah, false, bl, (short)0, c, false, false);
  c = __builtin_amdgcn_wmma_f32_16x16x32_bf16(false, al, false, bh, (short)0, c, false, false);
}

// fused SAGE layer: out = relu(agg @ Wl + bias + xin @ Wr), all 256x256
__global__ void __launch_bounds__(128)
sage_gemm_k(const float* __restrict__ agg, const float* __restrict__ xin,
            const __bf16* __restrict__ wlp, const __bf16* __restrict__ wrp,
            const float* __restrict__ bias, float* __restrict__ out) {
  const int K = 256, N = 256, NTN = 16;
  int wave   = threadIdx.x >> 5;
  int lane   = threadIdx.x & 31;
  int m_base = blockIdx.x << 4;
  int khalf  = (lane < 16) ? 0 : 8;
  int mrow   = m_base + (lane & 15);
  v8f c[4] = {};
  const float* arow = agg + (size_t)mrow * K;
  #pragma unroll 2
  for (int kt = 0; kt < 8; ++kt) {
    v16bf ah, al;
    load_a16(arow, (kt << 5) + khalf, ah, al);
    #pragma unroll
    for (int t = 0; t < 4; ++t) {
      int nt = (wave << 2) + t;
      wmma3(c[t], ah, al, wlp + (((size_t)(kt * NTN + nt) << 5) + lane) * 32);
    }
  }
  const float* brow = xin + (size_t)mrow * K;
  #pragma unroll 2
  for (int kt = 0; kt < 8; ++kt) {
    v16bf ah, al;
    load_a16(brow, (kt << 5) + khalf, ah, al);
    #pragma unroll
    for (int t = 0; t < 4; ++t) {
      int nt = (wave << 2) + t;
      wmma3(c[t], ah, al, wrp + (((size_t)(kt * NTN + nt) << 5) + lane) * 32);
    }
  }
  int rbase = m_base + ((lane < 16) ? 0 : 8);
  #pragma unroll
  for (int t = 0; t < 4; ++t) {
    int colx = (wave << 6) + (t << 4) + (lane & 15);
    float bv = bias[colx];
    #pragma unroll
    for (int r = 0; r < 8; ++r) {
      float v = c[t][r] + bv;
      out[(size_t)(rbase + r) * N + colx] = fmaxf(v, 0.0f);
    }
  }
}

// plain GEMM + bias (no activation); K=256, N in {256,128}; launch (N/64) waves
__global__ void gemm_bias_k(const float* __restrict__ A, const __bf16* __restrict__ Wp,
                            const float* __restrict__ bias, float* __restrict__ out,
                            int N) {
  const int K = 256;
  int ntn    = N >> 4;
  int wave   = threadIdx.x >> 5;
  int lane   = threadIdx.x & 31;
  int m_base = blockIdx.x << 4;
  int khalf  = (lane < 16) ? 0 : 8;
  int mrow   = m_base + (lane & 15);
  v8f c[4] = {};
  const float* arow = A + (size_t)mrow * K;
  #pragma unroll 2
  for (int kt = 0; kt < 8; ++kt) {
    v16bf ah, al;
    load_a16(arow, (kt << 5) + khalf, ah, al);
    #pragma unroll
    for (int t = 0; t < 4; ++t) {
      int nt = (wave << 2) + t;
      wmma3(c[t], ah, al, Wp + (((size_t)(kt * ntn + nt) << 5) + lane) * 32);
    }
  }
  int rbase = m_base + ((lane < 16) ? 0 : 8);
  #pragma unroll
  for (int t = 0; t < 4; ++t) {
    int colx = (wave << 6) + (t << 4) + (lane & 15);
    float bv = bias[colx];
    #pragma unroll
    for (int r = 0; r < 8; ++r) {
      out[(size_t)(rbase + r) * N + colx] = c[t][r] + bv;
    }
  }
}

// ======================= row-wise log_softmax over 128 cols =======================
__global__ void logsoftmax_k(const float* __restrict__ in, float* __restrict__ out) {
  int row = blockIdx.x * 8 + (threadIdx.x >> 5);
  if (row >= N_NODES) return;
  int lane = threadIdx.x & 31;
  const float* p = in + (size_t)row * D_OUTC;
  float v0 = p[lane], v1 = p[lane + 32], v2 = p[lane + 64], v3 = p[lane + 96];
  float m = fmaxf(fmaxf(v0, v1), fmaxf(v2, v3));
  #pragma unroll
  for (int off = 16; off > 0; off >>= 1) m = fmaxf(m, __shfl_xor(m, off, 32));
  float s = expf(v0 - m) + expf(v1 - m) + expf(v2 - m) + expf(v3 - m);
  #pragma unroll
  for (int off = 16; off > 0; off >>= 1) s += __shfl_xor(s, off, 32);
  float ls = m + logf(s);
  float* q = out + (size_t)row * D_OUTC;
  q[lane]      = v0 - ls;
  q[lane + 32] = v1 - ls;
  q[lane + 64] = v2 - ls;
  q[lane + 96] = v3 - ls;
}

// ======================= driver =======================
extern "C" void kernel_launch(void* const* d_in, const int* in_sizes, int n_in,
                              void* d_out, int out_size, void* d_ws, size_t ws_size,
                              hipStream_t stream) {
  (void)in_sizes; (void)n_in; (void)out_size; (void)ws_size;
  const float*     x       = (const float*)d_in[0];
  const long long* ei      = (const long long*)d_in[1];
  const float*     conv_wl = (const float*)d_in[2];
  const float*     conv_bl = (const float*)d_in[3];
  const float*     conv_wr = (const float*)d_in[4];
  const float*     mp_w1   = (const float*)d_in[5];
  const float*     mp_b1   = (const float*)d_in[6];
  const float*     mp_w2   = (const float*)d_in[7];
  const float*     mp_b2   = (const float*)d_in[8];
  float* out = (float*)d_out;

  char* ws = (char*)d_ws;
  size_t off = 0;
  auto carve = [&](size_t bytes) -> char* {
    char* p = ws + off;
    off += (bytes + 255) & ~(size_t)255;
    return p;
  };

  float* h1     = (float*)carve((size_t)N_NODES * D_HID * 4);
  float* h2     = (float*)carve((size_t)N_NODES * D_HID * 4);
  float* agg    = (float*)carve((size_t)N_NODES * D_HID * 4);
  float* logits = agg;  // reuse: agg is dead after last conv layer
  int* cnt      = (int*)carve((size_t)N_NODES * 4);
  int* row_ptr  = (int*)carve((size_t)N_NODES * 4);
  int* cursor   = (int*)carve((size_t)N_NODES * 4);
  int* excl     = (int*)carve((size_t)N_NODES * 4);
  int* bsum     = (int*)carve(256 * 4);
  int* col      = (int*)carve((size_t)N_EDGES * 4);
  const size_t PACK256 = (size_t)8 * 16 * 32 * 32;  // bf16 elems for 256x256 (hi+lo)
  const size_t PACK128 = (size_t)8 * 8 * 32 * 32;   // bf16 elems for 256x128
  __bf16* wlp = (__bf16*)carve(3 * PACK256 * 2);
  __bf16* wrp = (__bf16*)carve(3 * PACK256 * 2);
  __bf16* w1p = (__bf16*)carve(PACK256 * 2);
  __bf16* w2p = (__bf16*)carve(PACK128 * 2);

  // ---- pack all weights to WMMA bf16 hi/lo layout ----
  for (int i = 0; i < 3; ++i) {
    pack_w_k<<<16, 256, 0, stream>>>(conv_wl + (size_t)i * D_HID * D_HID,
                                     wlp + (size_t)i * PACK256, 256, 256);
    pack_w_k<<<16, 256, 0, stream>>>(conv_wr + (size_t)i * D_HID * D_HID,
                                     wrp + (size_t)i * PACK256, 256, 256);
  }
  pack_w_k<<<16, 256, 0, stream>>>(mp_w1, w1p, 256, 256);
  pack_w_k<<<8,  256, 0, stream>>>(mp_w2, w2p, 256, 128);

  // ---- build CSR (dst -> list of src) once; reused by all 3 layers ----
  const int NB = (N_NODES + 255) / 256;  // 196
  zero_i32_k   <<<NB, 256, 0, stream>>>(cnt, N_NODES);
  hist_k       <<<N_EDGES / 256, 256, 0, stream>>>(ei, cnt);
  scan_blocks_k<<<NB, 256, 0, stream>>>(cnt, excl, bsum, N_NODES);
  scan_small_k <<<1,  256, 0, stream>>>(bsum, NB);
  add_off_k    <<<NB, 256, 0, stream>>>(excl, bsum, row_ptr, cursor, N_NODES);
  fill_k       <<<N_EDGES / 256, 256, 0, stream>>>(ei, cursor, col);

  // ---- 3 SAGE layers ----
  const float* hcur = x;
  float* bufs[2] = {h1, h2};
  for (int i = 0; i < 3; ++i) {
    agg_mean_k<<<(N_NODES + 7) / 8, 256, 0, stream>>>(hcur, row_ptr, cnt, col, agg);
    float* hnext = bufs[i & 1];
    sage_gemm_k<<<N_NODES / 16, 128, 0, stream>>>(
        agg, hcur, wlp + (size_t)i * PACK256, wrp + (size_t)i * PACK256,
        conv_bl + (size_t)i * D_HID, hnext);
    hcur = hnext;
  }

  // ---- post-MP + log_softmax ----
  gemm_bias_k<<<N_NODES / 16, 128, 0, stream>>>(hcur, w1p, mp_b1, h2, 256);
  gemm_bias_k<<<N_NODES / 16, 64,  0, stream>>>(h2,   w2p, mp_b2, logits, 128);
  logsoftmax_k<<<(N_NODES + 7) / 8, 256, 0, stream>>>(logits, out);
}
